// Model_15315853377536
// MI455X (gfx1250) — compile-verified, matrix operand order checked by reading, not run on hardware
//
#include <hip/hip_runtime.h>
#include <hip/hip_bf16.h>

// ---------------------------------------------------------------------------
// Persistent single-kernel GRU rollout for MI455X (gfx1250).
// B=128, A=64, T=20, CTX=8, H=256 -> 8192 independent rows.
// 256 WGs x 256 threads; each WG owns 32 rows and iterates all 20 timesteps
// with the hidden state resident in LDS. bf16 WMMA GEMMs, f32 state carry.
// gi and gh GEMM passes run sequentially to halve accumulator pressure.
// ---------------------------------------------------------------------------

typedef __attribute__((ext_vector_type(16))) __bf16 v16bf;
typedef __attribute__((ext_vector_type(8)))  float  v8f;

union BF16x16 { uint4 u[2]; v16bf v; };

__device__ __forceinline__ v8f wmma_bf16(v16bf a, v16bf b, v8f c) {
    // D = A(16x32 bf16) * B(32x16 bf16) + C(16x16 f32)
    return __builtin_amdgcn_wmma_f32_16x16x32_bf16(
        false, a, false, b, (short)0, c, false, false);
}

// A fragment, row-major [16 x K], leading dim LD (bf16 elements).
// ISA 16-bit A layout: lanes 0-15 -> M=lane, K {0..7,16..23};
// lanes 16-31 -> M=lane-16, K {8..15,24..31}.
template <int LD>
__device__ __forceinline__ v16bf load_a(const __bf16* base, int lane, int kt) {
    const __bf16* p = base + (lane & 15) * LD + kt * 32 + (lane >> 4) * 8;
    BF16x16 t;
    t.u[0] = ((const uint4*)p)[0];
    t.u[1] = ((const uint4*)p)[2];
    return t.v;
}

// B fragment from pre-packed fragment-major buffer (512 bf16 per fragment).
__device__ __forceinline__ v16bf load_b(const __bf16* packed, int frag, int lane) {
    const __bf16* p = packed + frag * 512 + lane * 16;
    BF16x16 t;
    t.u[0] = ((const uint4*)p)[0];
    t.u[1] = ((const uint4*)p)[1];
    return t.v;
}

__device__ __forceinline__ float sigmoidf_(float x) {
    return 1.0f / (1.0f + __expf(-x));
}
__device__ __forceinline__ float eluf_(float x) {
    return x > 0.0f ? x : (__expf(x) - 1.0f);
}

// ---------------------------------------------------------------------------
// Prep: pack all weight matrices (row-major [N][K] fp32) into WMMA-B bf16
// fragments.  Fragment index = ntile*(K/32) + kt.
//   N = ntile*16 + (lane&15)
//   K = kt*32 + (v<4 ? 2v : 16+2(v-4)) + 8*(lane>>4) + (e&1),  v = e>>1
// Packed layout: wi (768x256) | wh (768x256) | fc1 (128x256) | fc2 (128x128)
// ---------------------------------------------------------------------------
__device__ __forceinline__ float pack_fetch(const float* W, int N, int K, int local) {
    int KT   = K >> 5;
    int f    = local >> 9;
    int r    = local & 511;
    int lane = r >> 4;
    int e    = r & 15;
    int ntile = f / KT;
    int kt    = f - ntile * KT;
    int v     = e >> 1;
    int k = kt * 32 + (v < 4 ? 2 * v : 16 + 2 * (v - 4)) + 8 * (lane >> 4) + (e & 1);
    int n = ntile * 16 + (lane & 15);
    return W[n * K + k];
}

__global__ void pack_all(const float* __restrict__ wi, const float* __restrict__ wh,
                         const float* __restrict__ f1, const float* __restrict__ f2,
                         __bf16* __restrict__ out) {
    int idx = blockIdx.x * 256 + threadIdx.x;   // 442368 packed elements
    float v;
    if (idx < 196608)       v = pack_fetch(wi, 768, 256, idx);
    else if (idx < 393216)  v = pack_fetch(wh, 768, 256, idx - 196608);
    else if (idx < 425984)  v = pack_fetch(f1, 128, 256, idx - 393216);
    else                    v = pack_fetch(f2, 128, 128, idx - 425984);
    out[idx] = (__bf16)v;
}

// ---------------------------------------------------------------------------
// Persistent rollout kernel.  grid=256, block=256 (8 waves).
// Wave w: mt = w&1 (16-row sub-tile), ng = w>>1 (192-col gate slice).
// ---------------------------------------------------------------------------
__global__ __launch_bounds__(256)
void gru_rollout(const float* __restrict__ inputs,           // [8192][20][2]
                 const unsigned char* __restrict__ eps_mask, // [12][64]
                 const float* __restrict__ embed_w,          // [256][2]
                 const float* __restrict__ embed_b,          // [256]
                 const float* __restrict__ gru_bi,           // [768]
                 const float* __restrict__ gru_bh,           // [768]
                 const float* __restrict__ fc1_b,            // [128]
                 const float* __restrict__ fc2_b,            // [128]
                 const float* __restrict__ fc3_w,            // [2][128]
                 const float* __restrict__ fc3_b,            // [2]
                 const __bf16* __restrict__ packed,          // packed weights
                 float* __restrict__ out)                    // [8192][12][2]
{
    __shared__ __align__(16) __bf16 s_inp[32][256];   // 16 KB  GRU input (bf16)
    __shared__ __align__(16) __bf16 s_h  [32][256];   // 16 KB  hidden (bf16)
    __shared__ __align__(16) float  s_hf [32][256];   // 32 KB  hidden (f32)
    __shared__ __align__(16) float  s_scr[32][1024];  // 128 KB gates / y1 / y2
    __shared__ float s_bi[768], s_bh[768];            //  6 KB
    __shared__ float s_f1b[128], s_f2b[128];          //  1 KB

    const int tid  = threadIdx.x;
    const int wave = tid >> 5;
    const int lane = tid & 31;
    const int mt   = wave & 1;        // M sub-tile (0/1)
    const int ng   = wave >> 1;       // gate N group (0..3) -> cols [ng*192, +192)
    const int m0   = blockIdx.x * 32; // first global row

    // ---- one-time init: zero hidden state, stage biases --------------------
    {
        int m  = tid >> 3;
        int jb = (tid & 7) * 32;
#pragma unroll
        for (int jj = 0; jj < 32; ++jj) {
            s_hf[m][jb + jj] = 0.0f;
            s_h [m][jb + jj] = (__bf16)0.0f;
        }
        for (int i = tid; i < 768; i += 256) { s_bi[i] = gru_bi[i]; s_bh[i] = gru_bh[i]; }
        if (tid < 128) { s_f1b[tid] = fc1_b[tid]; s_f2b[tid] = fc2_b[tid]; }
    }
    __syncthreads();

    for (int t = 0; t < 20; ++t) {
        // Launder a zero *offset* (not the pointers) so the t-invariant
        // B-fragment loads cannot be hoisted out of the timestep loop
        // (hoisting caused massive scratch spilling), while the base pointer
        // keeps its global address-space (global_load_b128, not flat_load).
        size_t zofs = 0;
        asm volatile("" : "+s"(zofs));
        const __bf16* wi_p = packed + zofs;
        const __bf16* wh_p = packed + zofs + 196608;
        const __bf16* f1_p = packed + zofs + 393216;
        const __bf16* f2_p = packed + zofs + 425984;

        // ---- phase 0: build GRU input tile (embedding / h feedback) -------
        {
            int m   = tid >> 3;
            int jb  = (tid & 7) * 32;
            int row = m0 + m;
            float in0 = inputs[(row * 20 + t) * 2 + 0];
            float in1 = inputs[(row * 20 + t) * 2 + 1];
            bool use_h = (t >= 8) && (eps_mask[(t - 8) * 64 + (row & 63)] != 0);
#pragma unroll
            for (int jj = 0; jj < 32; ++jj) {
                int j = jb + jj;
                float x = in0 * embed_w[j * 2 + 0] + in1 * embed_w[j * 2 + 1] + embed_b[j];
                s_inp[m][j] = (__bf16)(use_h ? s_hf[m][j] : x);
            }
        }
        __syncthreads();

        // ---- phase 1a: gi = inp @ Wi^T, stage to LDS ----------------------
        // s_scr row layout: [rz(512) | inn(256) | hn(256)]
        {
            v8f acc[12];
#pragma unroll
            for (int i = 0; i < 12; ++i) acc[i] = (v8f){};

            const __bf16* Ai = &s_inp[mt * 16][0];
#pragma unroll
            for (int kt = 0; kt < 8; ++kt) {
                v16bf a = load_a<256>(Ai, lane, kt);
#pragma unroll
                for (int i = 0; i < 12; ++i)
                    acc[i] = wmma_bf16(a, load_b(wi_p, (ng * 12 + i) * 8 + kt, lane), acc[i]);
            }
#pragma unroll
            for (int i = 0; i < 12; ++i) {
                int ncol = ng * 192 + i * 16 + (lane & 15);
#pragma unroll
                for (int v = 0; v < 8; ++v) {
                    int mm = mt * 16 + v + 8 * (lane >> 4);
                    s_scr[mm][ncol] = acc[i][v];   // rz region: gi; n region: inn
                }
            }
        }

        // ---- phase 1b: gh = h @ Wh^T, fold into LDS (same-thread elements)-
        {
            v8f acc[12];
#pragma unroll
            for (int i = 0; i < 12; ++i) acc[i] = (v8f){};

            const __bf16* Ah = &s_h[mt * 16][0];
#pragma unroll
            for (int kt = 0; kt < 8; ++kt) {
                v16bf a = load_a<256>(Ah, lane, kt);
#pragma unroll
                for (int i = 0; i < 12; ++i)
                    acc[i] = wmma_bf16(a, load_b(wh_p, (ng * 12 + i) * 8 + kt, lane), acc[i]);
            }
#pragma unroll
            for (int i = 0; i < 12; ++i) {
                int ncol = ng * 192 + i * 16 + (lane & 15);
#pragma unroll
                for (int v = 0; v < 8; ++v) {
                    int mm = mt * 16 + v + 8 * (lane >> 4);
                    float gh = acc[i][v];
                    if (ncol < 512) s_scr[mm][ncol] += gh;        // r,z: sum
                    else            s_scr[mm][ncol + 256] = gh;   // hn at 768+j
                }
            }
        }
        __syncthreads();

        // ---- phase 2: gate nonlinearity + hidden update -------------------
        {
            int m  = tid >> 3;
            int jb = (tid & 7) * 32;
#pragma unroll
            for (int jj = 0; jj < 32; ++jj) {
                int j = jb + jj;
                float r = sigmoidf_(s_scr[m][j]       + s_bi[j]       + s_bh[j]);
                float z = sigmoidf_(s_scr[m][256 + j] + s_bi[256 + j] + s_bh[256 + j]);
                float n = tanhf(s_scr[m][512 + j] + s_bi[512 + j] +
                                r * (s_scr[m][768 + j] + s_bh[512 + j]));
                float hp = s_hf[m][j];
                float hn = (1.0f - z) * n + z * hp;
                s_hf[m][j] = hn;
                s_h [m][j] = (__bf16)hn;
            }
        }
        __syncthreads();

        // ---- phase 3: fc1 [32x256]@[256x128] + elu -> y1 (bf16, s_scr) ----
        __bf16* y1 = (__bf16*)&s_scr[0][0];                      // [32][128], 8 KB
        float*  y2 = (float*)((char*)&s_scr[0][0] + 8 * 1024);   // [32][128], 16 KB
        {
            int nt = wave >> 1;                 // 0..3; tiles nt and nt+4
            v8f fa = (v8f){}, fb = (v8f){};
#pragma unroll
            for (int kt = 0; kt < 8; ++kt) {
                v16bf a = load_a<256>(&s_h[mt * 16][0], lane, kt);
                fa = wmma_bf16(a, load_b(f1_p, nt * 8 + kt, lane), fa);
                fb = wmma_bf16(a, load_b(f1_p, (nt + 4) * 8 + kt, lane), fb);
            }
            int n0 = nt * 16 + (lane & 15);
#pragma unroll
            for (int v = 0; v < 8; ++v) {
                int mm = mt * 16 + v + 8 * (lane >> 4);
                y1[mm * 128 + n0]      = (__bf16)eluf_(fa[v] + s_f1b[n0]);
                y1[mm * 128 + n0 + 64] = (__bf16)eluf_(fb[v] + s_f1b[n0 + 64]);
            }
        }
        __syncthreads();

        // ---- phase 4: fc2 [32x128]@[128x128] + elu -> y2 (f32) ------------
        {
            int nt = wave >> 1;
            v8f fa = (v8f){}, fb = (v8f){};
#pragma unroll
            for (int kt = 0; kt < 4; ++kt) {
                v16bf a = load_a<128>(y1 + mt * 16 * 128, lane, kt);
                fa = wmma_bf16(a, load_b(f2_p, nt * 4 + kt, lane), fa);
                fb = wmma_bf16(a, load_b(f2_p, (nt + 4) * 4 + kt, lane), fb);
            }
            int n0 = nt * 16 + (lane & 15);
#pragma unroll
            for (int v = 0; v < 8; ++v) {
                int mm = mt * 16 + v + 8 * (lane >> 4);
                y2[mm * 128 + n0]      = eluf_(fa[v] + s_f2b[n0]);      // disjoint
                y2[mm * 128 + n0 + 64] = eluf_(fb[v] + s_f2b[n0 + 64]); // from y1
            }
        }
        __syncthreads();

        // ---- phase 5: fc3 (N=2) + gt mask + output ------------------------
        if (t >= 8 && tid < 64) {
            int m = tid >> 1;
            int c = tid & 1;
            float s = fc3_b[c];
#pragma unroll 8
            for (int k = 0; k < 128; ++k)
                s += y2[m * 128 + k] * fc3_w[c * 128 + k];
            int row = m0 + m;
            float gt = inputs[(row * 20 + t) * 2 + c];
            out[row * 24 + (t - 8) * 2 + c] = (gt != 0.0f) ? s : 0.0f;
        }
        __syncthreads();   // protect s_scr reuse in next timestep
    }
}

// ---------------------------------------------------------------------------
// Host launcher: 2 launches total.
// ---------------------------------------------------------------------------
extern "C" void kernel_launch(void* const* d_in, const int* in_sizes, int n_in,
                              void* d_out, int out_size, void* d_ws, size_t ws_size,
                              hipStream_t stream) {
    (void)in_sizes; (void)n_in; (void)out_size; (void)ws_size;

    const float*         inputs   = (const float*)d_in[0];
    /* d_in[1] = context_frame_num (static 8, hardcoded) */
    const unsigned char* eps_mask = (const unsigned char*)d_in[2];
    const float*         embed_w  = (const float*)d_in[3];
    const float*         embed_b  = (const float*)d_in[4];
    const float*         gru_wi   = (const float*)d_in[5];
    const float*         gru_wh   = (const float*)d_in[6];
    const float*         gru_bi   = (const float*)d_in[7];
    const float*         gru_bh   = (const float*)d_in[8];
    const float*         fc1_w    = (const float*)d_in[9];
    const float*         fc1_b    = (const float*)d_in[10];
    const float*         fc2_w    = (const float*)d_in[11];
    const float*         fc2_b    = (const float*)d_in[12];
    const float*         fc3_w    = (const float*)d_in[13];
    const float*         fc3_b    = (const float*)d_in[14];

    __bf16* packed = (__bf16*)d_ws;   // 442368 bf16 = 864 KB

    pack_all<<<1728, 256, 0, stream>>>(gru_wi, gru_wh, fc1_w, fc2_w, packed);

    gru_rollout<<<256, 256, 0, stream>>>(inputs, eps_mask, embed_w, embed_b,
                                         gru_bi, gru_bh, fc1_b, fc2_b,
                                         fc3_w, fc3_b, packed, (float*)d_out);
}
